// SelfMatching_24026047054512
// MI455X (gfx1250) — matrix operand order repaired and checked by reading, not compiled, not run on hardware
//
#include <hip/hip_runtime.h>
#include <math.h>

#define DEV static __device__ __forceinline__

typedef __bf16 bf16_t;
typedef __attribute__((ext_vector_type(16))) __bf16 v16bf;
typedef __attribute__((ext_vector_type(8)))  __bf16 bf16x8;
typedef __attribute__((ext_vector_type(8)))  float   v8f;
typedef __attribute__((ext_vector_type(4)))  float   f32x4;

constexpr int B = 8, S = 2048, D = 256;
constexpr float VERY_LARGE = 1000000.0f;

// ---- WMMA operand loaders (16-bit, wave32 layouts per CDNA5 ISA 7.12.2) ----
// A 16x32: lane L holds row M=L&15; VGPR0..3 = K=(L>>4)*8+0..7, VGPR4..7 = K=16+(L>>4)*8+0..7
DEV v16bf load_A16(const bf16_t* base, int laneHi) {
  bf16x8 lo = *(const bf16x8*)(base + laneHi * 8);
  bf16x8 hi = *(const bf16x8*)(base + 16 + laneHi * 8);
  v16bf r;
#pragma unroll
  for (int i = 0; i < 8; ++i) { r[i] = lo[i]; r[i + 8] = hi[i]; }
  return r;
}
// B 32x16: lane n holds column N=n&15; K = (n>>4)*8+0..7, then 16+(n>>4)*8+0..7 (same as A)
DEV v16bf load_B16(const bf16_t* base, int laneHi) {
  bf16x8 lo = *(const bf16x8*)(base + laneHi * 8);
  bf16x8 hi = *(const bf16x8*)(base + 16 + laneHi * 8);
  v16bf r;
#pragma unroll
  for (int i = 0; i < 8; ++i) { r[i] = lo[i]; r[i + 8] = hi[i]; }
  return r;
}
DEV v8f wmma_bf16(v16bf a, v16bf b, v8f c) {
  return __builtin_amdgcn_wmma_f32_16x16x32_bf16(false, a, false, b, (short)0, c, false, false);
}

// ---- Kernel 1: prep. One wave per row: out[:,0:D]=input, xb=bf16(input),
// xw=bf16(input*w4mlu), q[row]=input.w4Q, mval[row]=(1-mask)*1e6.
// (sub0=input@w4C and bias are softmax-invariant -> dropped.)
__global__ __launch_bounds__(256) void prep_kernel(
    const float* __restrict__ input, const int* __restrict__ mask,
    const float* __restrict__ w4Q, const float* __restrict__ w4mlu,
    bf16_t* __restrict__ xb, bf16_t* __restrict__ xw,
    float* __restrict__ q, float* __restrict__ mval, float* __restrict__ out) {
  const int row  = blockIdx.x * 8 + (threadIdx.x >> 5);  // 0 .. B*S-1
  const int lane = threadIdx.x & 31;
  const int dd   = lane * 8;
  const float* xr = input + (size_t)row * D + dd;
  f32x4 x0 = *(const f32x4*)xr, x1 = *(const f32x4*)(xr + 4);

  // out column 0: raw input copy
  float* orow = out + (size_t)row * (6 * D) + dd;
  *(f32x4*)orow = x0; *(f32x4*)(orow + 4) = x1;

  f32x4 w0 = *(const f32x4*)(w4mlu + dd), w1 = *(const f32x4*)(w4mlu + dd + 4);
  f32x4 q0 = *(const f32x4*)(w4Q + dd),   q1 = *(const f32x4*)(w4Q + dd + 4);

  bf16x8 bx, bw;
  float part = 0.f;
#pragma unroll
  for (int i = 0; i < 4; ++i) {
    bx[i]     = (bf16_t)x0[i];           bx[i + 4] = (bf16_t)x1[i];
    bw[i]     = (bf16_t)(x0[i] * w0[i]); bw[i + 4] = (bf16_t)(x1[i] * w1[i]);
    part += x0[i] * q0[i] + x1[i] * q1[i];
  }
  *(bf16x8*)(xb + (size_t)row * D + dd) = bx;
  *(bf16x8*)(xw + (size_t)row * D + dd) = bw;
#pragma unroll
  for (int off = 16; off; off >>= 1) part += __shfl_xor(part, off);
  if (lane == 0) {
    q[row]    = part;
    mval[row] = (1.0f - (float)mask[row]) * VERY_LARGE;
  }
}

// ---- Kernel 2: batched LDS tile transpose, src [bz][R][C] -> dst [bz][C][R] (bf16 out)
template <typename SrcT>
__global__ void transpose_kernel(const SrcT* __restrict__ src, bf16_t* __restrict__ dst,
                                 int R, int Cc) {
  __shared__ bf16_t tile[32][33];
  const int bz = blockIdx.z;
  const int c0 = blockIdx.x * 32, r0 = blockIdx.y * 32;
  src += (size_t)bz * R * Cc;
  dst += (size_t)bz * R * Cc;
#pragma unroll
  for (int i = threadIdx.y; i < 32; i += 8)
    tile[i][threadIdx.x] = (bf16_t)(float)src[(size_t)(r0 + i) * Cc + c0 + threadIdx.x];
  __syncthreads();
#pragma unroll
  for (int i = threadIdx.y; i < 32; i += 8)
    dst[(size_t)(c0 + i) * R + r0 + threadIdx.x] = tile[threadIdx.x][i];
}

// ---- Kernel 3: scores + 2-pass flash softmax -> attn bf16 [B,S,S]
// One wave per 32-row s-group (2 strips). C tile = [t-rows x s-cols]:
//   A = xb rows t (streamed, shared by both strips), B = xw rows s (resident, 128 VGPRs).
// Lane n covers s = s0+16r+(n&15), t = t0+(n>>4)*8+0..7 -> contiguous attn stores.
__global__ __launch_bounds__(256) void score_kernel(
    const bf16_t* __restrict__ xb, const bf16_t* __restrict__ xw,
    const float* __restrict__ q, const float* __restrict__ mval,
    bf16_t* __restrict__ attn) {
  const int lane   = threadIdx.x & 31;
  const int grp    = blockIdx.x * 8 + (threadIdx.x >> 5);  // 0 .. B*S/32-1
  const int b      = grp / (S / 32);
  const int s0     = (grp % (S / 32)) * 32;
  const int laneLo = lane & 15, laneHi = lane >> 4;

  // resident B operands: xw rows s0..s0+31, full D (2 strips x 8 k-steps)
  v16bf Breg[2][8];
#pragma unroll
  for (int r = 0; r < 2; ++r) {
    const bf16_t* xwrow = xw + ((size_t)b * S + s0 + 16 * r + laneLo) * D;
#pragma unroll
    for (int kk = 0; kk < 8; ++kk) Breg[r][kk] = load_B16(xwrow + kk * 32, laneHi);
  }

  const bf16_t* xbB = xb + (size_t)b * S * D;
  const float*  qB  = q + b * S;
  const float*  mB  = mval + b * S;
  float ms[2] = {mB[s0 + laneLo], mB[s0 + 16 + laneLo]};

  float rmax[2] = {-INFINITY, -INFINITY}, rsum[2] = {0.0f, 0.0f};

  // ---- pass 1: online row max / sum-exp ----
  for (int t0 = 0; t0 < S; t0 += 16) {
    const bf16_t* arow = xbB + (size_t)(t0 + laneLo) * D;
    v8f c[2] = {};
#pragma unroll
    for (int kk = 0; kk < 8; ++kk) {
      v16bf a = load_A16(arow + kk * 32, laneHi);
      c[0] = wmma_bf16(a, Breg[0][kk], c[0]);
      c[1] = wmma_bf16(a, Breg[1][kk], c[1]);
    }
    const int tb = t0 + laneHi * 8;
    f32x4 qv0 = *(const f32x4*)(qB + tb), qv1 = *(const f32x4*)(qB + tb + 4);
    f32x4 mv0 = *(const f32x4*)(mB + tb), mv1 = *(const f32x4*)(mB + tb + 4);
#pragma unroll
    for (int st = 0; st < 2; ++st) {
      float v[8];
#pragma unroll
      for (int r = 0; r < 4; ++r) {
        v[r]     = c[st][r]     + qv0[r] - ms[st] * mv0[r];
        v[r + 4] = c[st][r + 4] + qv1[r] - ms[st] * mv1[r];
      }
      float tmax = v[0];
#pragma unroll
      for (int r = 1; r < 8; ++r) tmax = fmaxf(tmax, v[r]);
      const float mnew = fmaxf(rmax[st], tmax);
      float add = 0.f;
#pragma unroll
      for (int r = 0; r < 8; ++r) add += __expf(v[r] - mnew);
      rsum[st] = rsum[st] * __expf(rmax[st] - mnew) + add;
      rmax[st] = mnew;
    }
  }
  // merge the two lane-halves of each row (lane n <-> n+16)
  float gmax[2], ginv[2];
#pragma unroll
  for (int st = 0; st < 2; ++st) {
    const float omax = __shfl_xor(rmax[st], 16);
    const float osum = __shfl_xor(rsum[st], 16);
    gmax[st] = fmaxf(rmax[st], omax);
    ginv[st] = 1.0f / (rsum[st] * __expf(rmax[st] - gmax[st]) +
                       osum * __expf(omax - gmax[st]));
  }

  // ---- pass 2: recompute, normalize, store attn (contiguous b128 per lane) ----
  bf16_t* attnRow0 = attn + ((size_t)b * S + s0 + laneLo) * S;
  bf16_t* attnRow1 = attnRow0 + (size_t)16 * S;
  for (int t0 = 0; t0 < S; t0 += 16) {
    const bf16_t* arow = xbB + (size_t)(t0 + laneLo) * D;
    v8f c[2] = {};
#pragma unroll
    for (int kk = 0; kk < 8; ++kk) {
      v16bf a = load_A16(arow + kk * 32, laneHi);
      c[0] = wmma_bf16(a, Breg[0][kk], c[0]);
      c[1] = wmma_bf16(a, Breg[1][kk], c[1]);
    }
    const int tb = t0 + laneHi * 8;
    f32x4 qv0 = *(const f32x4*)(qB + tb), qv1 = *(const f32x4*)(qB + tb + 4);
    f32x4 mv0 = *(const f32x4*)(mB + tb), mv1 = *(const f32x4*)(mB + tb + 4);
#pragma unroll
    for (int st = 0; st < 2; ++st) {
      bf16x8 p;
#pragma unroll
      for (int r = 0; r < 4; ++r) {
        p[r]     = (bf16_t)(__expf(c[st][r]     + qv0[r] - ms[st] * mv0[r] - gmax[st]) * ginv[st]);
        p[r + 4] = (bf16_t)(__expf(c[st][r + 4] + qv1[r] - ms[st] * mv1[r] - gmax[st]) * ginv[st]);
      }
      *(bf16x8*)((st ? attnRow1 : attnRow0) + tb) = p;
    }
  }
}

// ---- Kernels 4/6: out-tile GEMM.  C[d-rows, s-cols] = sum_t AT[d,t] * attn[s,t]
//   A = AT rows d (xbT for mode 0, wT for mode 1), B = attn rows s.
// One wave computes 16 s x 64 d (4 d-tiles): B tile loaded once per k-step,
// reused by 4 WMMAs. Lane n holds row s=s0+(n&15), 8 consecutive d per tile.
__global__ __launch_bounds__(256) void gemm_kernel(
    const bf16_t* __restrict__ AT,    // [B][D][S]
    const bf16_t* __restrict__ attn,  // [B][S][S]
    const float* __restrict__ x,      // [B][S][D]
    float* __restrict__ out,          // [B][S][6D]
    bf16_t* __restrict__ wbuf,        // mode 0 only: weighted bf16 [B][S][D]
    int mode) {
  const int lane     = threadIdx.x & 31;
  const int tilesPB  = (S / 16) * (D / 64);                // 512
  const int tile     = blockIdx.x * 8 + (threadIdx.x >> 5);
  const int b        = tile / tilesPB;
  const int rem      = tile % tilesPB;
  const int s0       = (rem >> 2) * 16;
  const int d0       = (rem & 3) * 64;
  const int laneLo   = lane & 15, laneHi = lane >> 4;

  const bf16_t* Ab = AT + (size_t)b * D * S + (size_t)(d0 + laneLo) * S;
  const bf16_t* Bb = attn + ((size_t)b * S + s0 + laneLo) * S;

  v8f c[4] = {};
#pragma unroll 2
  for (int t = 0; t < S; t += 32) {
    v16bf bop = load_B16(Bb + t, laneHi);
#pragma unroll
    for (int j = 0; j < 4; ++j)
      c[j] = wmma_bf16(load_A16(Ab + (size_t)(j * 16) * S + t, laneHi), bop, c[j]);
  }

  const int s = s0 + laneLo;
  const float* xrow = x + ((size_t)b * S + s) * D;
  float* orow = out + ((size_t)b * S + s) * (6 * D);
#pragma unroll
  for (int j = 0; j < 4; ++j) {
    const int dd = d0 + j * 16 + laneHi * 8;
    f32x4 x0 = *(const f32x4*)(xrow + dd), x1 = *(const f32x4*)(xrow + dd + 4);
    f32x4 c0 = {c[j][0], c[j][1], c[j][2], c[j][3]};
    f32x4 c1 = {c[j][4], c[j][5], c[j][6], c[j][7]};
    if (mode == 0) {  // weighted, input*weighted, input-weighted
      *(f32x4*)(orow + 1 * D + dd) = c0;      *(f32x4*)(orow + 1 * D + dd + 4) = c1;
      *(f32x4*)(orow + 2 * D + dd) = x0 * c0; *(f32x4*)(orow + 2 * D + dd + 4) = x1 * c1;
      *(f32x4*)(orow + 3 * D + dd) = x0 - c0; *(f32x4*)(orow + 3 * D + dd + 4) = x1 - c1;
      bf16x8 wb;
#pragma unroll
      for (int i = 0; i < 4; ++i) { wb[i] = (bf16_t)c0[i]; wb[i + 4] = (bf16_t)c1[i]; }
      *(bf16x8*)(wbuf + ((size_t)b * S + s) * D + dd) = wb;
    } else {          // twotime, input-twotime
      *(f32x4*)(orow + 4 * D + dd) = c0;      *(f32x4*)(orow + 4 * D + dd + 4) = c1;
      *(f32x4*)(orow + 5 * D + dd) = x0 - c0; *(f32x4*)(orow + 5 * D + dd + 4) = x1 - c1;
    }
  }
}

extern "C" void kernel_launch(void* const* d_in, const int* in_sizes, int n_in,
                              void* d_out, int out_size, void* d_ws, size_t ws_size,
                              hipStream_t stream) {
  const float* input  = (const float*)d_in[0];
  const int*   mask   = (const int*)d_in[1];
  // d_in[2] = w4C (softmax-invariant, unused), d_in[5] = bias (unused)
  const float* w4Q    = (const float*)d_in[3];
  const float* w4mlu  = (const float*)d_in[4];
  float* out = (float*)d_out;

  // workspace carve-up
  char* w = (char*)d_ws;
  const size_t nBSD = (size_t)B * S * D;     // 4,194,304 elems
  bf16_t* xb   = (bf16_t*)w;                w += nBSD * 2;
  bf16_t* xw   = (bf16_t*)w;                w += nBSD * 2;
  bf16_t* xbT  = (bf16_t*)w;                w += nBSD * 2;
  bf16_t* wbuf = (bf16_t*)w;                w += nBSD * 2;
  bf16_t* wT   = (bf16_t*)w;                w += nBSD * 2;
  float*  q    = (float*)w;                 w += (size_t)B * S * 4;
  float*  mval = (float*)w;                 w += (size_t)B * S * 4;
  bf16_t* attn = (bf16_t*)w;                // B*S*S*2 = 64 MiB

  // 1. prep: bf16 converts, q, mval, out col 0
  prep_kernel<<<(B * S) / 8, 256, 0, stream>>>(input, mask, w4Q, w4mlu,
                                               xb, xw, q, mval, out);
  // 2. input (f32) -> xbT (bf16, [B][D][S])
  transpose_kernel<float><<<dim3(D / 32, S / 32, B), dim3(32, 8), 0, stream>>>(
      input, xbT, S, D);
  // 3. scores + softmax -> attn bf16 (2 s-strips per wave)
  score_kernel<<<(B * (S / 32)) / 8, 256, 0, stream>>>(xb, xw, q, mval, attn);
  // 4. weighted = attn @ input ; out cols 1-3 ; wbuf bf16
  gemm_kernel<<<(B * (S / 16) * (D / 64)) / 8, 256, 0, stream>>>(
      xbT, attn, input, out, wbuf, 0);
  // 5. wbuf (bf16) -> wT (bf16, [B][D][S])
  transpose_kernel<bf16_t><<<dim3(D / 32, S / 32, B), dim3(32, 8), 0, stream>>>(
      wbuf, wT, S, D);
  // 6. twotime = attn @ weighted ; out cols 4-5
  gemm_kernel<<<(B * (S / 16) * (D / 64)) / 8, 256, 0, stream>>>(
      wT, attn, input, out, nullptr, 1);
}
